// QKVAttentionLayer_42880953483476
// MI455X (gfx1250) — matrix operand order repaired and checked by reading, not compiled, not run on hardware
//
#include <hip/hip_runtime.h>
#include <hip/hip_bf16.h>

// MI455X (gfx1250) sigmoid-attention, bf16 WMMA 16x16x32 (f32 acc), wave32.
//   q = X Wq^T + bq ; v = X Wv^T + bv ; k = v Wk^T + bk   (keys from projected values!)
//   P = sigmoid(mask ? q k^T : -1e9) ; out = P v
// ws (bf16): Q[16384x768] | K | V | Vt[8][768][2048] | P[8][2048][2048]  (~168 MB)

typedef __bf16 bf16_t;
typedef __bf16 v16bf __attribute__((ext_vector_type(16)));
typedef __bf16 v8bf  __attribute__((ext_vector_type(8)));
typedef float  v8f   __attribute__((ext_vector_type(8)));

#define BM 128
#define BN 128
#define BK 32
#define LDT 40      // LDS row stride (bf16 elems): 80B, 16B-aligned, bank-padded
#define LSEQ 2048   // sequence length (fixed by the reference)

#if defined(__gfx1250__) && __has_builtin(__builtin_amdgcn_global_load_async_to_lds_b128)
#define USE_ASYNC 1
// builtin signature (from hipcc diagnostic): (int4 addrspace(1)*, int4 addrspace(3)*, imm, imm)
typedef int v4i_gcc __attribute__((vector_size(16)));
typedef __attribute__((address_space(1))) v4i_gcc as1_v4i;
typedef __attribute__((address_space(3))) v4i_gcc as3_v4i;
#else
#define USE_ASYNC 0
#endif

__device__ __forceinline__ void wait_async_zero() {
#if USE_ASYNC
#if __has_builtin(__builtin_amdgcn_s_wait_asynccnt)
  __builtin_amdgcn_s_wait_asynccnt(0);
#else
  asm volatile("s_wait_asynccnt 0x0" ::: "memory");
#endif
#endif
}

// ---- fragment loaders (CDNA5 ISA 7.12.2 VGPR layouts) ----------------------
// A 16x32 bf16: lane L -> row M=L&15; lanes<16: K {0..7,16..23}, lanes>=16: {8..15,24..31}
__device__ __forceinline__ v16bf load_a_frag(const bf16_t* s, int m0, int lane) {
  const bf16_t* p = s + (m0 + (lane & 15)) * LDT + ((lane >> 4) << 3);
  v8bf lo = *(const v8bf*)p;
  v8bf hi = *(const v8bf*)(p + 16);
  v16bf a;
#pragma unroll
  for (int i = 0; i < 8; ++i) { a[i] = lo[i]; a[i + 8] = hi[i]; }
  return a;
}
// B 32x16 bf16: lane L -> col N=L&15; lanes<16: K {0..7,16..23}, lanes>=16: {8..15,24..31}
__device__ __forceinline__ v16bf load_b_frag(const bf16_t* s, int n0, int lane) {
  const bf16_t* p = s + (n0 + (lane & 15)) * LDT + ((lane >> 4) << 3);
  v8bf lo = *(const v8bf*)p;
  v8bf hi = *(const v8bf*)(p + 16);
  v16bf b;
#pragma unroll
  for (int i = 0; i < 8; ++i) { b[i] = lo[i]; b[i + 8] = hi[i]; }
  return b;
}

__device__ __forceinline__ void wmma_tile(const bf16_t* As, const bf16_t* Bs,
                                          v8f acc[4][2], int wm, int wn, int lane) {
  v16bf bfr[2];
#pragma unroll
  for (int j = 0; j < 2; ++j) bfr[j] = load_b_frag(Bs, wn + 16 * j, lane);
#pragma unroll
  for (int i = 0; i < 4; ++i) {
    v16bf afr = load_a_frag(As, wm + 16 * i, lane);
#pragma unroll
    for (int j = 0; j < 2; ++j)
      acc[i][j] = __builtin_amdgcn_wmma_f32_16x16x32_bf16(
          false, afr, false, bfr[j], (short)0, acc[i][j], false, false);
  }
}

// ---- tile stagers: 128 rows x 32 cols, 256 threads --------------------------
// bf16 copy: async DMA straight into LDS (no VGPR round trip) when available.
__device__ __forceinline__ void issue_bf16(const bf16_t* g, int ldg, bf16_t* s,
                                           v8bf* regs, int tid) {
#pragma unroll
  for (int i = 0; i < 2; ++i) {
    int idx = tid + 256 * i;
    int row = idx >> 2, col = (idx & 3) << 3;
    const bf16_t* gp = g + (size_t)row * ldg + col;
#if USE_ASYNC
    (void)regs;
    __builtin_amdgcn_global_load_async_to_lds_b128(
        (as1_v4i*)gp, (as3_v4i*)(s + row * LDT + col), 0, 0);
#else
    (void)s;
    regs[i] = *(const v8bf*)gp;
#endif
  }
}
__device__ __forceinline__ void commit_bf16(const v8bf* regs, bf16_t* s, int tid) {
#if !USE_ASYNC
#pragma unroll
  for (int i = 0; i < 2; ++i) {
    int idx = tid + 256 * i;
    int row = idx >> 2, col = (idx & 3) << 3;
    *(v8bf*)(s + row * LDT + col) = regs[i];
  }
#else
  (void)regs; (void)s; (void)tid;
#endif
}
// f32 -> bf16 (needs VGPR pass-through for the convert): split issue/commit so the
// loads for tile t+1 stay in flight across the WMMA work on tile t.
__device__ __forceinline__ void issue_f32(const float* g, int ldg, float4* regs, int tid) {
#pragma unroll
  for (int i = 0; i < 4; ++i) {
    int idx = tid + 256 * i;
    int row = idx >> 3, col = (idx & 7) << 2;
    regs[i] = *(const float4*)(g + (size_t)row * ldg + col);
  }
}
__device__ __forceinline__ void commit_f32(const float4* regs, bf16_t* s, int tid) {
#pragma unroll
  for (int i = 0; i < 4; ++i) {
    int idx = tid + 256 * i;
    int row = idx >> 3, col = (idx & 7) << 2;
    bf16_t* d = s + row * LDT + col;
    d[0] = (bf16_t)regs[i].x; d[1] = (bf16_t)regs[i].y;
    d[2] = (bf16_t)regs[i].z; d[3] = (bf16_t)regs[i].w;
  }
}

// A-operand dispatch by element type (overload instead of if-constexpr)
__device__ __forceinline__ void issue_a(const float* g, int ldg, bf16_t* s,
                                        float4* fr, v8bf* br, int tid) {
  (void)s; (void)br; issue_f32(g, ldg, fr, tid);
}
__device__ __forceinline__ void issue_a(const bf16_t* g, int ldg, bf16_t* s,
                                        float4* fr, v8bf* br, int tid) {
  (void)fr; issue_bf16(g, ldg, s, br, tid);
}
__device__ __forceinline__ void commit_a(const float*, const float4* fr, const v8bf* br,
                                         bf16_t* s, int tid) { (void)br; commit_f32(fr, s, tid); }
__device__ __forceinline__ void commit_a(const bf16_t*, const float4* fr, const v8bf* br,
                                         bf16_t* s, int tid) { (void)fr; commit_bf16(br, s, tid); }

// ---- pipelined bf16 x bf16 core (score / context) ---------------------------
__device__ __forceinline__ void gemm_core_bf16(const bf16_t* gA, int lda,
                                               const bf16_t* gB, int ldb, int Kdim,
                                               bf16_t (*As)[BM * LDT], bf16_t (*Bs)[BN * LDT],
                                               v8f acc[4][2], int tid, int wm, int wn, int lane) {
  v8bf ra[2], rb[2];
  issue_bf16(gA, lda, As[0], ra, tid);
  issue_bf16(gB, ldb, Bs[0], rb, tid);
  const int nk = Kdim / BK;
  for (int t = 0; t < nk; ++t) {
    const int cur = t & 1;
    commit_bf16(ra, As[cur], tid);
    commit_bf16(rb, Bs[cur], tid);
    wait_async_zero();
    __syncthreads();
    if (t + 1 < nk) {
      issue_bf16(gA + (t + 1) * BK, lda, As[cur ^ 1], ra, tid);
      issue_bf16(gB + (t + 1) * BK, ldb, Bs[cur ^ 1], rb, tid);
    }
    wmma_tile(As[cur], Bs[cur], acc, wm, wn, lane);
  }
}

// 1) projection: C_bf16[m,n] = sum_k A[m,k] W[n,k] + bias[n]; optional transposed copy
template <typename AT>
__global__ void __launch_bounds__(256)
proj_gemm_kernel(const AT* __restrict__ A, const float* __restrict__ W,
                 const float* __restrict__ bias, bf16_t* __restrict__ C,
                 bf16_t* __restrict__ Ct, int N, int Kdim) {
  __shared__ bf16_t As[2][BM * LDT];
  __shared__ bf16_t Bs[2][BN * LDT];
  const int tid = threadIdx.x, lane = tid & 31, wid = tid >> 5;
  const int wm = (wid >> 2) * 64, wn = (wid & 3) * 32;
  const int M0 = blockIdx.x * BM, N0 = blockIdx.y * BN;
  const AT* gA = A + (size_t)M0 * Kdim;
  const float* gW = W + (size_t)N0 * Kdim;
  v8f acc[4][2] = {};

  float4 fa[4]; v8bf ba[2]; float4 fb[4];
  issue_a(gA, Kdim, As[0], fa, ba, tid);
  issue_f32(gW, Kdim, fb, tid);
  const int nk = Kdim / BK;
  for (int t = 0; t < nk; ++t) {
    const int cur = t & 1;
    commit_a(gA, fa, ba, As[cur], tid);
    commit_f32(fb, Bs[cur], tid);
    wait_async_zero();
    __syncthreads();
    if (t + 1 < nk) {
      issue_a(gA + (t + 1) * BK, Kdim, As[cur ^ 1], fa, ba, tid);
      issue_f32(gW + (t + 1) * BK, Kdim, fb, tid);
    }
    wmma_tile(As[cur], Bs[cur], acc, wm, wn, lane);
  }

#pragma unroll
  for (int i = 0; i < 4; ++i)
#pragma unroll
    for (int j = 0; j < 2; ++j) {
      const int n = N0 + wn + 16 * j + (lane & 15);
      const float bv = bias[n];
      const int mb = M0 + wm + 16 * i + ((lane >> 4) << 3);  // C layout: vgpr v -> M = v (+8 hi)
#pragma unroll
      for (int v = 0; v < 8; ++v) {
        const int m = mb + v;
        const bf16_t h = (bf16_t)(acc[i][j][v] + bv);
        C[(size_t)m * N + n] = h;
        if (Ct) {  // transposed copy [b][n][l] for the async context-B operand
          const int bI = m >> 11, l = m & (LSEQ - 1);
          Ct[((size_t)bI * N + n) * LSEQ + l] = h;
        }
      }
    }
}

// 2) score: P[q,kk] = sigmoid(mask ? Q[q,:].K[kk,:] : -inf) -> bf16
__global__ void __launch_bounds__(256)
score_kernel(const bf16_t* __restrict__ Q, const bf16_t* __restrict__ Km,
             const unsigned char* __restrict__ mask, bf16_t* __restrict__ P,
             int L, int D) {
  const int b = blockIdx.z;
  Q += (size_t)b * L * D; Km += (size_t)b * L * D;
  mask += (size_t)b * L * L; P += (size_t)b * L * L;
  __shared__ bf16_t As[2][BM * LDT];
  __shared__ bf16_t Bs[2][BN * LDT];
  const int tid = threadIdx.x, lane = tid & 31, wid = tid >> 5;
  const int wm = (wid >> 2) * 64, wn = (wid & 3) * 32;
  const int M0 = blockIdx.x * BM, N0 = blockIdx.y * BN;
  v8f acc[4][2] = {};
  gemm_core_bf16(Q + (size_t)M0 * D, D, Km + (size_t)N0 * D, D, D,
                 As, Bs, acc, tid, wm, wn, lane);

#pragma unroll
  for (int i = 0; i < 4; ++i)
#pragma unroll
    for (int j = 0; j < 2; ++j) {
      const int n = N0 + wn + 16 * j + (lane & 15);
      const int mb = M0 + wm + 16 * i + ((lane >> 4) << 3);
#pragma unroll
      for (int v = 0; v < 8; ++v) {
        const size_t off = (size_t)(mb + v) * L + n;
        const float s = acc[i][j][v];
        const float p = mask[off] ? (1.0f / (1.0f + __expf(-s))) : 0.0f;
        P[off] = (bf16_t)p;
      }
    }
}

// 3) context: O_f32[q,d] = sum_kk P[q,kk] * Vt[d,kk]
__global__ void __launch_bounds__(256)
context_kernel(const bf16_t* __restrict__ P, const bf16_t* __restrict__ Vt,
               float* __restrict__ O, int L, int D) {
  const int b = blockIdx.z;
  P += (size_t)b * L * L; Vt += (size_t)b * D * L; O += (size_t)b * L * D;
  __shared__ bf16_t As[2][BM * LDT];
  __shared__ bf16_t Bs[2][BN * LDT];
  const int tid = threadIdx.x, lane = tid & 31, wid = tid >> 5;
  const int wm = (wid >> 2) * 64, wn = (wid & 3) * 32;
  const int M0 = blockIdx.x * BM, N0 = blockIdx.y * BN;
  v8f acc[4][2] = {};
  gemm_core_bf16(P + (size_t)M0 * L, L, Vt + (size_t)N0 * L, L, L,
                 As, Bs, acc, tid, wm, wn, lane);

#pragma unroll
  for (int i = 0; i < 4; ++i)
#pragma unroll
    for (int j = 0; j < 2; ++j) {
      const int n = N0 + wn + 16 * j + (lane & 15);
      const int mb = M0 + wm + 16 * i + ((lane >> 4) << 3);
#pragma unroll
      for (int v = 0; v < 8; ++v)
        O[(size_t)(mb + v) * D + n] = acc[i][j][v];
    }
}

extern "C" void kernel_launch(void* const* d_in, const int* in_sizes, int n_in,
                              void* d_out, int out_size, void* d_ws, size_t ws_size,
                              hipStream_t stream) {
  (void)in_sizes; (void)n_in; (void)out_size; (void)ws_size;
  const int B = 8, L = LSEQ, D = 768;

  const float*         queries = (const float*)d_in[0];
  const float*         values  = (const float*)d_in[1];
  const unsigned char* mask    = (const unsigned char*)d_in[2];  // jax bool = 1 byte
  const float* Wq = (const float*)d_in[3];
  const float* bq = (const float*)d_in[4];
  const float* Wk = (const float*)d_in[5];
  const float* bk = (const float*)d_in[6];
  const float* Wv = (const float*)d_in[7];
  const float* bv = (const float*)d_in[8];
  float* out = (float*)d_out;

  bf16_t* Qb  = (bf16_t*)d_ws;
  bf16_t* Kb  = Qb  + (size_t)B * L * D;
  bf16_t* Vb  = Kb  + (size_t)B * L * D;
  bf16_t* Vtb = Vb  + (size_t)B * L * D;   // [B][D][L]
  bf16_t* Pb  = Vtb + (size_t)B * L * D;   // [B][L][L]

  const dim3 blk(256);
  const dim3 gproj((B * L) / BM, D / BN);
  proj_gemm_kernel<float><<<gproj, blk, 0, stream>>>(queries, Wq, bq, Qb, nullptr, D, D);
  proj_gemm_kernel<float><<<gproj, blk, 0, stream>>>(values,  Wv, bv, Vb, Vtb,     D, D);
  proj_gemm_kernel<bf16_t><<<gproj, blk, 0, stream>>>(Vb, Wk, bk, Kb, nullptr, D, D);

  const dim3 gscore(L / BM, L / BN, B);
  score_kernel<<<gscore, blk, 0, stream>>>(Qb, Kb, mask, Pb, L, D);

  const dim3 gctx(L / BM, D / BN, B);
  context_kernel<<<gctx, blk, 0, stream>>>(Pb, Vtb, out, L, D);
}
